// SAGClassifer_79843442033168
// MI455X (gfx1250) — matrix operand order repaired and checked by reading, not compiled, not run on hardware
//
#include <hip/hip_runtime.h>
#include <hip/hip_bf16.h>

// CDNA5 / gfx1250 implementation of the SAGPool GNN reference.
// Matrix work: v_wmma_f32_16x16x32_bf16 (bf16 in, f32 accumulate).
// Operand feed: 16-byte vector loads only (rows zero-padded to 16, weights
// pre-transposed so B-operand K-runs are contiguous). EXEC all-1s in the
// WMMA kernel (no divergent guards), per ISA 7.12 requirement.

typedef __bf16 v16bf __attribute__((ext_vector_type(16)));
typedef __bf16 v8bf  __attribute__((ext_vector_type(8)));
typedef float  v8f   __attribute__((ext_vector_type(8)));

#define NODES0 20000
#define HID    128

static __device__ __forceinline__ v16bf cat16(v8bf lo, v8bf hi) {
  return __builtin_shufflevector(lo, hi, 0, 1, 2, 3, 4, 5, 6, 7,
                                 8, 9, 10, 11, 12, 13, 14, 15);
}

// -------------------------------------------- f32 -> bf16 with zero row-pad
__global__ void f32_to_bf16_pad_kernel(const float* __restrict__ in,
                                       __bf16* __restrict__ out,
                                       int count, int total) {
  int i = blockIdx.x * blockDim.x + threadIdx.x;
  if (i >= total) return;
  out[i] = (i < count) ? (__bf16)in[i] : (__bf16)0.0f;
}

// ------------------------------- weight transpose + bf16 (one-time, 98 KB)
// WT[l][n][k] = W[l][k][n]: makes the WMMA B-operand K-runs contiguous.
__global__ void transpose_bf16_kernel(const float* __restrict__ W,
                                      __bf16* __restrict__ WT, int total) {
  int idx = blockIdx.x * blockDim.x + threadIdx.x;
  if (idx >= total) return;
  int l = idx / (HID * HID);
  int r = (idx / HID) & 127;   // output row  (= original column)
  int c = idx & 127;           // output col  (= original row)
  WT[idx] = (__bf16)W[l * HID * HID + c * HID + r];
}

// ------------------------------------------------- mean aggregation -> bf16
// dst = repeat(arange(n), k): node i owns edge slots [i*k, (i+1)*k), deg == k.
// Rows [n, npad) are zero-filled for the WMMA tiles.
__global__ void mean_agg_bf16_kernel(const float* __restrict__ x,
                                     const int* __restrict__ src,
                                     __bf16* __restrict__ aggbf,
                                     int n, int k, int total) {
  int idx = blockIdx.x * blockDim.x + threadIdx.x;
  if (idx >= total) return;
  int i = idx >> 7, c = idx & 127;
  if (i >= n) { aggbf[idx] = (__bf16)0.0f; return; }
  float s = 0.0f;
  for (int e = 0; e < k; ++e) s += x[src[i * k + e] * HID + c];
  aggbf[idx] = (__bf16)(s / (float)k);
}

// ------------------------------------------------------- WMMA GraphConv GEMM
// out[npad x 128] = relu(Abf @ Wrel + Xbf @ Wroot + bias).
// grid = npad/16 blocks of 256 threads (8 waves = 8 column tiles).
// Per 16x16 tile: 4 K-steps x 2 WMMA; all operand loads are b128.
__global__ void conv_wmma_kernel(const __bf16* __restrict__ Abf,
                                 const __bf16* __restrict__ Xbf,
                                 const __bf16* __restrict__ WrelT,
                                 const __bf16* __restrict__ WrootT,
                                 const float* __restrict__ bias,
                                 float* __restrict__ out) {
  const int lane  = threadIdx.x & 31;
  const int wtile = threadIdx.x >> 5;   // column tile 0..7
  const int m0    = blockIdx.x * 16;
  const int half  = lane >> 4;          // 0: lanes 0-15, 1: lanes 16-31
  const int l16   = lane & 15;
  // A: lane owns row m0+l16; elems 0..7 -> K=ka.., 8..15 -> K=ka+16..
  const __bf16* ap = Abf + (size_t)(m0 + l16) * HID + half * 8;
  const __bf16* xp = Xbf + (size_t)(m0 + l16) * HID + half * 8;
  // B (transposed weights): lane owns column bcol; elems 0..7 -> K=kb.., 8..15 -> K=kb+16..
  const int bcol = wtile * 16 + l16;
  const __bf16* bp = WrelT  + (size_t)bcol * HID + half * 8;
  const __bf16* cp = WrootT + (size_t)bcol * HID + half * 8;

  v8f c = {};
#pragma unroll
  for (int k0 = 0; k0 < HID; k0 += 32) {
    v16bf a  = cat16(*(const v8bf*)(ap + k0), *(const v8bf*)(ap + k0 + 16));
    v16bf xv = cat16(*(const v8bf*)(xp + k0), *(const v8bf*)(xp + k0 + 16));
    v16bf br = cat16(*(const v8bf*)(bp + k0), *(const v8bf*)(bp + k0 + 16));
    v16bf bo = cat16(*(const v8bf*)(cp + k0), *(const v8bf*)(cp + k0 + 16));
    c = __builtin_amdgcn_wmma_f32_16x16x32_bf16(false, a,  false, br,
                                                (short)0, c, false, false);
    c = __builtin_amdgcn_wmma_f32_16x16x32_bf16(false, xv, false, bo,
                                                (short)0, c, false, false);
  }
  const float bv = bias[bcol];
#pragma unroll
  for (int r = 0; r < 8; ++r)  // C/D: VGPR r -> row r (lanes<16) / r+8
    out[(size_t)(m0 + half * 8 + r) * HID + bcol] = fmaxf(c[r] + bv, 0.0f);
}

// --------------------------------------------- per-row dots with Wprel/Wproot
__global__ void row_dots_kernel(const float* __restrict__ x,
                                const float* __restrict__ wprel,
                                const float* __restrict__ wproot,
                                float* __restrict__ t, float* __restrict__ s,
                                int n) {
  int row  = blockIdx.x * (blockDim.x >> 5) + (threadIdx.x >> 5);
  int lane = threadIdx.x & 31;
  if (row >= n) return;
  float a = 0.0f, b = 0.0f;
  for (int c = lane; c < HID; c += 32) {
    float v = x[row * HID + c];
    a += v * wprel[c];
    b += v * wproot[c];
  }
  for (int off = 16; off; off >>= 1) {
    a += __shfl_down(a, off, 32);
    b += __shfl_down(b, off, 32);
  }
  if (lane == 0) { t[row] = a; s[row] = b; }
}

// score_i = tanh( sum_e t[src[i*k+e]] + s[i] + bprel )
__global__ void score_kernel(const float* __restrict__ t,
                             const float* __restrict__ s,
                             const int* __restrict__ src, int n, int k,
                             const float* __restrict__ bprel,
                             float* __restrict__ score) {
  int i = blockIdx.x * blockDim.x + threadIdx.x;
  if (i >= n) return;
  float a = s[i] + *bprel;
  for (int e = 0; e < k; ++e) a += t[src[i * k + e]];
  score[i] = tanhf(a);
}

// ------------------------------------- exact lax.top_k via 64-bit bitonic sort
// key = (descending-monotone(score) << 32) | index  => ascending sort yields
// descending score, ties broken by ascending index (XLA top_k semantics).
__global__ void build_keys_kernel(const float* __restrict__ score,
                                  unsigned long long* __restrict__ keys,
                                  int n, int size) {
  int i = blockIdx.x * blockDim.x + threadIdx.x;
  if (i >= size) return;
  if (i < n) {
    unsigned u = __float_as_uint(score[i]);
    unsigned m = (u & 0x80000000u) ? ~u : (u | 0x80000000u);  // ascending map
    keys[i] = ((unsigned long long)(~m) << 32) | (unsigned)i; // descending
  } else {
    keys[i] = ~0ull;  // padding sorts to the end
  }
}

// global pass (j > 256)
__global__ void bitonic_step_kernel(unsigned long long* __restrict__ keys,
                                    int j, int kk, int size) {
  int i = blockIdx.x * blockDim.x + threadIdx.x;
  if (i >= size) return;
  int ixj = i ^ j;
  if (ixj <= i) return;
  unsigned long long a = keys[i], b = keys[ixj];
  bool up = ((i & kk) == 0);
  if (up ? (a > b) : (a < b)) { keys[i] = b; keys[ixj] = a; }
}

// fused LDS passes: all j = jstart..1 (jstart <= 256) within 512-elem segments
__global__ void bitonic_local_kernel(unsigned long long* __restrict__ keys,
                                     int kk, int jstart) {
  __shared__ unsigned long long sh[512];
  const int base = blockIdx.x * 512;
  const int tid  = threadIdx.x;
  sh[tid]       = keys[base + tid];
  sh[tid + 256] = keys[base + tid + 256];
  __syncthreads();
  for (int j = jstart; j > 0; j >>= 1) {
    int i  = ((tid & ~(j - 1)) << 1) | (tid & (j - 1));  // pair (i, i+j)
    int gi = base + i;
    bool up = ((gi & kk) == 0);
    unsigned long long a = sh[i], b = sh[i + j];
    if (up ? (a > b) : (a < b)) { sh[i] = b; sh[i + j] = a; }
    __syncthreads();
  }
  keys[base + tid]       = sh[tid];
  keys[base + tid + 256] = sh[tid + 256];
}

__global__ void extract_topk_kernel(const unsigned long long* __restrict__ keys,
                                    const float* __restrict__ score,
                                    int* __restrict__ perm,
                                    float* __restrict__ vals, int kkeep) {
  int i = blockIdx.x * blockDim.x + threadIdx.x;
  if (i >= kkeep) return;
  int p = (int)(keys[i] & 0xffffffffu);
  perm[i] = p;
  vals[i] = score[p];
}

// x_out[r] = x_in[perm[r]] * vals[r]; optionally permute pos too.
__global__ void gather_kernel(const float* __restrict__ xin,
                              const float* __restrict__ posin,
                              const int* __restrict__ perm,
                              const float* __restrict__ vals,
                              float* __restrict__ xout,
                              float* __restrict__ posout, int nk, int do_pos) {
  int idx = blockIdx.x * blockDim.x + threadIdx.x;
  if (idx >= nk * HID) return;
  int r = idx >> 7, c = idx & 127;
  int p = perm[r];
  xout[idx] = xin[p * HID + c] * vals[r];
  if (do_pos && c < 2) posout[r * 2 + c] = posin[p * 2 + c];
}

// global_max_pool: per-column max over n rows
__global__ void colmax_kernel(const float* __restrict__ x, int n,
                              float* __restrict__ xs) {
  __shared__ float red[256];
  int col = blockIdx.x, tid = threadIdx.x;
  float m = -INFINITY;
  for (int r = tid; r < n; r += 256) m = fmaxf(m, x[r * HID + col]);
  red[tid] = m;
  __syncthreads();
  for (int s2 = 128; s2; s2 >>= 1) {
    if (tid < s2) red[tid] = fmaxf(red[tid], red[tid + s2]);
    __syncthreads();
  }
  if (tid == 0) xs[col] = red[0];
}

// ------------------------------------------------ KNN (k nearest, loop=False)
// Strict-< insertion while scanning j ascending reproduces top_k(-d2) order:
// ascending d2, ties broken by lower index.
template <int K>
__global__ void knn_kernel(const float* __restrict__ pos, int n,
                           int* __restrict__ nbr) {
  int i = blockIdx.x * blockDim.x + threadIdx.x;
  if (i >= n) return;
  float px = pos[2 * i], py = pos[2 * i + 1];
  float bd[K]; int bi[K];
#pragma unroll
  for (int e = 0; e < K; ++e) { bd[e] = INFINITY; bi[e] = 0; }
  for (int j = 0; j < n; ++j) {
    if (j == i) continue;                       // diagonal = inf in reference
    float dx = px - pos[2 * j], dy = py - pos[2 * j + 1];
    float d2 = dx * dx + dy * dy;
    if (d2 < bd[K - 1]) {
      bd[K - 1] = d2; bi[K - 1] = j;
#pragma unroll
      for (int p = K - 1; p > 0; --p) {         // bubble-up (rest stays sorted)
        if (bd[p] < bd[p - 1]) {
          float td = bd[p]; bd[p] = bd[p - 1]; bd[p - 1] = td;
          int   ti = bi[p]; bi[p] = bi[p - 1]; bi[p - 1] = ti;
        }
      }
    }
  }
#pragma unroll
  for (int e = 0; e < K; ++e) nbr[i * K + e] = bi[e];
}

// ----------------------------------------------------------- final MLP (tiny)
__global__ void mlp_kernel(const float* __restrict__ xs,
                           const float* __restrict__ W1,
                           const float* __restrict__ b1,
                           const float* __restrict__ W2,
                           const float* __restrict__ b2,
                           float* __restrict__ out) {
  __shared__ float h2[HID];
  int c = threadIdx.x;  // 128 threads
  float acc = b1[c];
  for (int k = 0; k < 3 * HID; ++k) acc += xs[k] * W1[k * HID + c];
  h2[c] = fmaxf(acc, 0.0f);
  __syncthreads();
  if (c < 5) {
    float o = b2[c];
    for (int q = 0; q < HID; ++q) o += h2[q] * W2[q * 5 + c];
    out[c] = o;
  }
}

// ============================================================================
extern "C" void kernel_launch(void* const* d_in, const int* in_sizes, int n_in,
                              void* d_out, int out_size, void* d_ws,
                              size_t ws_size, hipStream_t stream) {
  (void)in_sizes; (void)n_in; (void)out_size; (void)ws_size;
  const float* x_in    = (const float*)d_in[0];
  const float* pos_in  = (const float*)d_in[1];
  const int*   edge    = (const int*)d_in[2];   // [2,E]; first E ints = src row
  const float* Wrel    = (const float*)d_in[3];
  const float* brel    = (const float*)d_in[4];
  const float* Wroot   = (const float*)d_in[5];
  const float* Wprel   = (const float*)d_in[6];
  const float* bprel   = (const float*)d_in[7];
  const float* Wproot  = (const float*)d_in[8];
  const float* W1      = (const float*)d_in[9];
  const float* b1      = (const float*)d_in[10];
  const float* W2      = (const float*)d_in[11];
  const float* b2      = (const float*)d_in[12];
  float* out = (float*)d_out;

  // ---- workspace carve-out (all static; ~33 MB total) ----
  char* ws = (char*)d_ws;
  size_t off = 0;
  auto carve = [&](size_t bytes) -> void* {
    void* p = (void*)(ws + off);
    off = (off + bytes + 255) & ~(size_t)255;
    return p;
  };
  float*  XA      = (float*)carve((size_t)NODES0 * HID * 4);  // pooled x
  float*  XC      = (float*)carve((size_t)NODES0 * HID * 4);  // conv output
  __bf16* ABF     = (__bf16*)carve((size_t)NODES0 * HID * 2);
  __bf16* XBF     = (__bf16*)carve((size_t)NODES0 * HID * 2);
  __bf16* WRELT   = (__bf16*)carve((size_t)3 * HID * HID * 2);
  __bf16* WROOTT  = (__bf16*)carve((size_t)3 * HID * HID * 2);
  float*  T       = (float*)carve((size_t)NODES0 * 4);
  float*  S       = (float*)carve((size_t)NODES0 * 4);
  float*  SC      = (float*)carve((size_t)NODES0 * 4);
  unsigned long long* KEYS = (unsigned long long*)carve((size_t)32768 * 8);
  int*    PERM    = (int*)carve((size_t)10000 * 4);
  float*  VALS    = (float*)carve((size_t)10000 * 4);
  float*  PA      = (float*)carve((size_t)NODES0 * 2 * 4);
  float*  PB      = (float*)carve((size_t)NODES0 * 2 * 4);
  int*    NBR     = (int*)carve((size_t)10000 * 8 * 4);
  float*  XS      = (float*)carve((size_t)3 * HID * 4);

  // ---- one-time weight transpose + bf16 conversion ----
  {
    int cnt = 3 * HID * HID;
    transpose_bf16_kernel<<<(cnt + 255) / 256, 256, 0, stream>>>(Wrel,  WRELT,  cnt);
    transpose_bf16_kernel<<<(cnt + 255) / 256, 256, 0, stream>>>(Wroot, WROOTT, cnt);
  }

  int n = NODES0;
  const float* xcur   = x_in;
  const float* poscur = pos_in;
  const int*   srccur = edge;  // layer0: k=6, node i owns slots [6i,6i+6)
  int kdeg = 6;

  for (int layer = 0; layer < 3; ++layer) {
    const int npad = (n + 15) & ~15;       // zero-padded rows for WMMA tiles
    const int nc = n * HID, ncp = npad * HID;
    // bf16 operands: x and mean-aggregated neighbors (rows [n,npad) zeroed)
    f32_to_bf16_pad_kernel<<<(ncp + 255) / 256, 256, 0, stream>>>(xcur, XBF, nc, ncp);
    mean_agg_bf16_kernel<<<(ncp + 255) / 256, 256, 0, stream>>>(
        xcur, srccur, ABF, n, kdeg, ncp);
    // GraphConv via WMMA: XC = relu(ABF@Wrel + XBF@Wroot + brel)
    conv_wmma_kernel<<<npad / 16, 256, 0, stream>>>(
        ABF, XBF, WRELT + (size_t)layer * HID * HID,
        WROOTT + (size_t)layer * HID * HID, brel + layer * HID, XC);
    // scoring head
    row_dots_kernel<<<(n + 7) / 8, 256, 0, stream>>>(
        XC, Wprel + layer * HID, Wproot + layer * HID, T, S, n);
    score_kernel<<<(n + 255) / 256, 256, 0, stream>>>(
        T, S, srccur, n, kdeg, bprel + layer, SC);
    // exact top_k: bitonic sort of (score,index) keys; low-j passes fused in LDS
    int size = 1; while (size < n) size <<= 1;
    build_keys_kernel<<<(size + 255) / 256, 256, 0, stream>>>(SC, KEYS, n, size);
    for (int kk = 2; kk <= size; kk <<= 1) {
      int j = kk >> 1;
      for (; j > 256; j >>= 1)
        bitonic_step_kernel<<<(size + 255) / 256, 256, 0, stream>>>(KEYS, j, kk, size);
      bitonic_local_kernel<<<size / 512, 256, 0, stream>>>(KEYS, kk, j);
    }
    int kkeep = (n + 1) / 2;
    extract_topk_kernel<<<(kkeep + 255) / 256, 256, 0, stream>>>(KEYS, SC, PERM, VALS, kkeep);
    // pooled x (and pos for layers that still need KNN)
    float* posdst = (layer == 0) ? PA : PB;
    int do_pos = (layer < 2) ? 1 : 0;
    gather_kernel<<<(kkeep * HID + 255) / 256, 256, 0, stream>>>(
        XC, poscur, PERM, VALS, XA, posdst, kkeep, do_pos);
    // JumpingKnowledge: global max pool of pooled x
    colmax_kernel<<<HID, 256, 0, stream>>>(XA, kkeep, XS + layer * HID);
    // rebuild KNN graph for next layer (the post-layer-2 KNN is dead code)
    if (layer == 0) {
      knn_kernel<6><<<(kkeep + 255) / 256, 256, 0, stream>>>(posdst, kkeep, NBR);
      srccur = NBR; kdeg = 6;
    } else if (layer == 1) {
      knn_kernel<8><<<(kkeep + 255) / 256, 256, 0, stream>>>(posdst, kkeep, NBR);
      srccur = NBR; kdeg = 8;
    }
    n = kkeep;
    xcur = XA;
    poscur = posdst;
  }

  mlp_kernel<<<1, HID, 0, stream>>>(XS, W1, b1, W2, b2, out);
}
